// DigitCaps_34428457845507
// MI455X (gfx1250) — matrix operand order repaired and checked by reading, not compiled, not run on hardware
//
#include <hip/hip_runtime.h>

// ---------------------------------------------------------------------------
// CapsNet dynamic routing, fused into bf16-WMMA GEMMs on gfx1250 (MI455X).
//   x: [B=512, I=1152, J=8] f32,  W: [I,O=10,D=16,J] f32,  out: [B,10,16,1] f32
// u_hat (377MB) is never materialized. Heavy contractions run as dense GEMMs
// on v_wmma_f32_16x16x32_bf16 (f32 accumulate). s-GEMM stages its shared B
// operand through LDS with GLOBAL_LOAD_ASYNC_TO_LDS_B128 (double-buffered,
// 2 k-chunks per barrier, ASYNCcnt-synchronized, last iteration peeled and
// unrolling suppressed so accumulators chain in place with no copies/NOPs);
// agreement GEMM uses 2x2 register blocking.
// ---------------------------------------------------------------------------

typedef __attribute__((ext_vector_type(16))) __bf16 v16bf;
typedef __attribute__((ext_vector_type(8)))  __bf16 v8bf;
typedef __attribute__((ext_vector_type(8)))  float  v8f;

#define B_    512
#define I_    1152
#define J_    8
#define O_    10
#define D_    16
#define K1    (I_ * J_)      // 9216  (contraction dim of s-GEMM)
#define N1    (O_ * D_)      // 160
#define KC1   (K1 / 32)      // 288 k-chunks of 32 for bf16 WMMA
#define MT1   (B_ / 16)      // 32 M tiles
#define NT1   (N1 / 16)      // 10 N tiles
#define KSPLIT 8
#define KC1P  (KC1 / KSPLIT) // 36 k-chunks per split
#define PAIRS (KC1P / 2)     // 18 double-chunk steps per split
#define NT2   (K1 / 16)      // 576 N tiles of agreement GEMM
#define KC2   (B_ / 32)      // 16 k-chunks (K=512)

// ---------------- utility -------------------------------------------------
__global__ void zero_f32(float* __restrict__ p, int n) {
    int t = blockIdx.x * blockDim.x + threadIdx.x;
    if (t < n) p[t] = 0.0f;
}

// ---------------- x -> A-layout tiles (16x32 bf16 per-lane swizzle) -------
// xA[mT][kc][lane][16]: lane&15 = row-in-tile, lane>>4 selects K sub-group.
// Per ISA 7.12.2: elems 0..7 -> K=kbase+0..7, elems 8..15 -> K=kbase+16..23.
__global__ void convert_xA(const float* __restrict__ x, __bf16* __restrict__ xA) {
    int tid = blockIdx.x * blockDim.x + threadIdx.x;
    if (tid >= MT1 * KC1 * 32) return;
    int lane = tid & 31;
    int kc   = (tid >> 5) % KC1;
    int mT   = tid / (32 * KC1);
    int b     = mT * 16 + (lane & 15);
    int kbase = kc * 32 + (lane >> 4) * 8;
    const float* src = x + (size_t)b * K1 + kbase;
    __bf16* dst = xA + (size_t)tid * 16;
#pragma unroll
    for (int t = 0; t < 8; ++t) dst[t]     = (__bf16)src[t];
#pragma unroll
    for (int t = 0; t < 8; ++t) dst[8 + t] = (__bf16)src[16 + t];
}

// ---------------- x -> B-layout tiles (32x16 bf16 per-lane swizzle) -------
// xB[nT][kc][lane][16]: lane&15 = col, elems 0..7 -> K=kbase+0..7, 8..15 -> K=kbase+16..23 (kbase=(lane>>4)*8).
__global__ void convert_xB(const float* __restrict__ x, __bf16* __restrict__ xB) {
    int tid = blockIdx.x * blockDim.x + threadIdx.x;
    if (tid >= NT2 * KC2 * 32) return;
    int lane = tid & 31;
    int kc   = (tid >> 5) % KC2;
    int nT   = tid / (32 * KC2);
    int col = lane & 15;
    int b0  = kc * 32 + (lane >> 4) * 8;
    __bf16* dst = xB + (size_t)tid * 16;
#pragma unroll
    for (int t = 0; t < 16; ++t)
        dst[t] = (__bf16)x[(size_t)(b0 + (t < 8 ? t : t + 8)) * K1 + nT * 16 + col];
}

// ---------------- softmax over input-capsule axis (one wave per o) --------
__global__ void softmax_c(const float* __restrict__ bij, float* __restrict__ c) {
    int o = threadIdx.x >> 5;       // blockDim = 320 -> 10 waves, o = wave id
    int lane = threadIdx.x & 31;
    float mx = -3.4e38f;
    for (int i = lane; i < I_; i += 32) mx = fmaxf(mx, bij[i * O_ + o]);
#pragma unroll
    for (int d = 16; d >= 1; d >>= 1) mx = fmaxf(mx, __shfl_xor(mx, d, 32));
    float sum = 0.0f;
    for (int i = lane; i < I_; i += 32) sum += __expf(bij[i * O_ + o] - mx);
#pragma unroll
    for (int d = 16; d >= 1; d >>= 1) sum += __shfl_xor(sum, d, 32);
    float inv = 1.0f / sum;
    for (int i = lane; i < I_; i += 32)
        c[i * O_ + o] = __expf(bij[i * O_ + o] - mx) * inv;
}

// ---------------- Wc[k=(i,j)][n=(o,d)] = c[i,o]*W[i,o,d,j], B-swizzled ----
__global__ void build_wc(const float* __restrict__ W, const float* __restrict__ c,
                         __bf16* __restrict__ wcb) {
    int tid = blockIdx.x * blockDim.x + threadIdx.x;
    if (tid >= NT1 * KC1 * 32) return;
    int lane = tid & 31;
    int kc   = (tid >> 5) % KC1;
    int nT   = tid / (32 * KC1);
    int o = nT;               // n = nT*16 + col, col<16 -> o = nT
    int d = lane & 15;
    int k0 = kc * 32 + (lane >> 4) * 8;
    __bf16* dst = wcb + (size_t)tid * 16;
#pragma unroll
    for (int t = 0; t < 16; ++t) {
        int k = k0 + (t < 8 ? t : t + 8), i = k >> 3, j = k & 7;
        float w = W[(((size_t)i * O_ + o) * D_ + d) * J_ + j];
        dst[t] = (__bf16)(w * c[i * O_ + o]);
    }
}

// ---------------- s-GEMM: s[b,od] = sum_k xA[b,k] * Wc[k,od] --------------
// M=512 N=160 K=9216. Block = 8 waves; two Wc B-tiles x two k-chunks (4KB)
// are staged per step into a double-buffered LDS region by ALL 256 threads
// (16B each, branch-free) via async-to-LDS loads with strength-reduced
// addresses. Each wave owns one M tile and both N tiles (4 WMMAs/step,
// interleaved acc0/acc1 chains). Last step peeled; unroll suppressed so the
// loop PHI pins accumulators -> in-place D==C WMMA, no copies, no NOPs.
// K split 8-way, flushed with f32 atomics.
__global__ void __launch_bounds__(256)
gemm_s(const __bf16* __restrict__ xA, const __bf16* __restrict__ wcb,
       float* __restrict__ s) {
    __shared__ char bstage[2][4096];  // [buf][nt*2048 + chunk*1024 + lane*32]
    const int tid  = threadIdx.x;
    const int lane = tid & 31;
    const int w    = tid >> 5;                // wave 0..7
    const int mT   = blockIdx.x * 8 + w;      // 32 M tiles / 8 waves
    const int nT0  = blockIdx.y * 2;          // N tile pair
    const int kc0  = blockIdx.z * KC1P;       // even

    const __bf16* aP = xA + (((size_t)mT * KC1 + kc0) * 32 + lane) * 16;

    // per-thread async staging: thread copies 16B; tiles are contiguous 2KB
    // regions in the pre-swizzled layout, so src just advances by 2048/step.
    const int nt = tid >> 7, c = tid & 127;
    unsigned long long src = (unsigned long long)(uintptr_t)
        ((const char*)wcb + ((size_t)(nT0 + nt) * KC1 + kc0) * 1024 + c * 16);
    const unsigned int dst0 =
        (unsigned int)(uintptr_t)&bstage[0][nt * 2048 + c * 16];

    v8f acc0 = {}, acc1 = {};

    auto consume = [&](int p) {
        v16bf a0 = *(const v16bf*)aP;          // k-chunk 2p
        v16bf a1 = *(const v16bf*)(aP + 512);  // k-chunk 2p+1
        aP += 1024;
        const v16bf* bb = (const v16bf*)&bstage[p & 1][0];
        v16bf b00 = bb[lane];                  // nT0,   chunk 2p
        v16bf b10 = bb[64 + lane];             // nT0+1, chunk 2p
        acc0 = __builtin_amdgcn_wmma_f32_16x16x32_bf16(
            false, a0, false, b00, (short)0, acc0, false, false);
        acc1 = __builtin_amdgcn_wmma_f32_16x16x32_bf16(
            false, a0, false, b10, (short)0, acc1, false, false);
        v16bf b01 = bb[32 + lane];             // nT0,   chunk 2p+1
        v16bf b11 = bb[96 + lane];             // nT0+1, chunk 2p+1
        acc0 = __builtin_amdgcn_wmma_f32_16x16x32_bf16(
            false, a1, false, b01, (short)0, acc0, false, false);
        acc1 = __builtin_amdgcn_wmma_f32_16x16x32_bf16(
            false, a1, false, b11, (short)0, acc1, false, false);
    };

    // prologue: stage step 0 into buffer 0
    asm volatile("global_load_async_to_lds_b128 %0, %1, off"
                 :: "v"(dst0), "v"(src) : "memory");
    src += 2048;

    // steady state: wait, barrier, stage next (unconditional), consume.
    // unroll 1: single body => loop PHI keeps accumulators in fixed VGPRs.
#pragma unroll 1
    for (int p = 0; p < PAIRS - 1; ++p) {
        asm volatile("s_wait_asynccnt 0x0" ::: "memory");  // staged data landed
        __syncthreads();
        unsigned int dst = dst0 ^ (((p + 1) & 1) ? 4096u : 0u);
        asm volatile("global_load_async_to_lds_b128 %0, %1, off"
                     :: "v"(dst), "v"(src) : "memory");
        src += 2048;
        consume(p);
    }
    // peeled final step: nothing left to stage
    asm volatile("s_wait_asynccnt 0x0" ::: "memory");
    __syncthreads();
    consume(PAIRS - 1);

    int col = lane & 15, rb = (lane >> 4) * 8;
    float* o0 = s + ((size_t)mT * 16 + rb) * N1 + nT0 * 16 + col;
#pragma unroll
    for (int r = 0; r < 8; ++r) atomicAdd(o0 + (size_t)r * N1, acc0[r]);
    float* o1 = o0 + 16;
#pragma unroll
    for (int r = 0; r < 8; ++r) atomicAdd(o1 + (size_t)r * N1, acc1[r]);
}

// ---------------- squash + transpose v for agreement GEMM -----------------
__global__ void squash_k(const float* __restrict__ s, __bf16* __restrict__ vT,
                         float* __restrict__ out, int writeOut) {
    int tid = blockIdx.x * blockDim.x + threadIdx.x;  // (b,o)
    if (tid >= B_ * O_) return;
    int b = tid / O_, o = tid % O_;
    const float* sp = s + (size_t)b * N1 + o * 16;
    float tmp[16], sq = 0.0f;
#pragma unroll
    for (int d = 0; d < 16; ++d) { tmp[d] = sp[d]; sq += tmp[d] * tmp[d]; }
    float scale = (sq / (1.0f + sq)) * rsqrtf(sq);
#pragma unroll
    for (int d = 0; d < 16; ++d) {
        float v = tmp[d] * scale;
        vT[(size_t)(o * 16 + d) * B_ + b] = (__bf16)v;   // [od][b] for GEMM A
        if (writeOut) out[(size_t)b * N1 + o * 16 + d] = v;
    }
}

// ---------------- agreement GEMM: M2[od,ij] = (1/B) sum_b v[b,od]x[b,ij] --
// M=160 N=9216 K=512; 2x2 register blocking: 4 WMMAs per 4 tile-loads.
__global__ void __launch_bounds__(32)
gemm_agree(const __bf16* __restrict__ vT, const __bf16* __restrict__ xB,
           float* __restrict__ M2) {
    int nT0 = blockIdx.x * 2;   // 0..574 over (i,j)
    int mT0 = blockIdx.y * 2;   // 0..8   over (o,d)
    int lane = threadIdx.x;
    int r0   = mT0 * 16 + (lane & 15);
    int r1   = r0 + 16;
    int kgrp = lane >> 4;
    const __bf16* b0P = xB + ((size_t)nT0 * KC2 * 32 + lane) * 16;
    const __bf16* b1P = xB + ((size_t)(nT0 + 1) * KC2 * 32 + lane) * 16;
    v8f acc00 = {}, acc01 = {}, acc10 = {}, acc11 = {};
#pragma unroll 2
    for (int kc = 0; kc < KC2; ++kc) {
        int kb = kc * 32 + kgrp * 8;
        v8bf a0lo = *(const v8bf*)(vT + (size_t)r0 * B_ + kb);
        v8bf a0hi = *(const v8bf*)(vT + (size_t)r0 * B_ + kb + 16);
        v8bf a1lo = *(const v8bf*)(vT + (size_t)r1 * B_ + kb);
        v8bf a1hi = *(const v8bf*)(vT + (size_t)r1 * B_ + kb + 16);
        v16bf a0, a1;
#pragma unroll
        for (int t = 0; t < 8; ++t) {
            a0[t] = a0lo[t]; a0[8 + t] = a0hi[t];
            a1[t] = a1lo[t]; a1[8 + t] = a1hi[t];
        }
        v16bf b0 = *(const v16bf*)b0P; b0P += 32 * 16;
        v16bf b1 = *(const v16bf*)b1P; b1P += 32 * 16;
        acc00 = __builtin_amdgcn_wmma_f32_16x16x32_bf16(
            false, a0, false, b0, (short)0, acc00, false, false);
        acc10 = __builtin_amdgcn_wmma_f32_16x16x32_bf16(
            false, a1, false, b0, (short)0, acc10, false, false);
        acc01 = __builtin_amdgcn_wmma_f32_16x16x32_bf16(
            false, a0, false, b1, (short)0, acc01, false, false);
        acc11 = __builtin_amdgcn_wmma_f32_16x16x32_bf16(
            false, a1, false, b1, (short)0, acc11, false, false);
    }
    int col = lane & 15, rb = (lane >> 4) * 8;
    const float invB = 1.0f / (float)B_;
    float* m00 = M2 + ((size_t)mT0 * 16 + rb) * K1 + nT0 * 16 + col;
#pragma unroll
    for (int r = 0; r < 8; ++r) {
        m00[(size_t)r * K1]             = acc00[r] * invB;   // (mT0,  nT0)
        m00[(size_t)r * K1 + 16]        = acc01[r] * invB;   // (mT0,  nT0+1)
        m00[((size_t)r + 16) * K1]      = acc10[r] * invB;   // (mT0+1,nT0)
        m00[((size_t)r + 16) * K1 + 16] = acc11[r] * invB;   // (mT0+1,nT0+1)
    }
}

// ---------------- b_ij += sum_{d,j} W[i,o,d,j] * M2[od, ij] ---------------
__global__ void bupdate(const float* __restrict__ W, const float* __restrict__ M2,
                        float* __restrict__ bij) {
    int tid = blockIdx.x * blockDim.x + threadIdx.x;  // (i,o)
    if (tid >= I_ * O_) return;
    int i = tid / O_, o = tid % O_;
    const float* wp = W + ((size_t)i * O_ + o) * (size_t)(D_ * J_);
    float sum = 0.0f;
#pragma unroll
    for (int d = 0; d < 16; ++d) {
        const float* mp = M2 + (size_t)(o * 16 + d) * K1 + i * 8;
#pragma unroll
        for (int j = 0; j < 8; ++j) sum += wp[d * 8 + j] * mp[j];
    }
    bij[tid] += sum;
}

// ---------------------------------------------------------------------------
extern "C" void kernel_launch(void* const* d_in, const int* in_sizes, int n_in,
                              void* d_out, int out_size, void* d_ws, size_t ws_size,
                              hipStream_t stream) {
    (void)in_sizes; (void)n_in; (void)out_size; (void)ws_size;
    const float* x = (const float*)d_in[0];
    const float* W = (const float*)d_in[1];
    float* out = (float*)d_out;

    char* ws = (char*)d_ws;
    size_t off = 0;
    auto carve = [&](size_t bytes) -> char* {
        char* p = ws + off;
        off = (off + bytes + 255) & ~(size_t)255;
        return p;
    };
    __bf16* xA   = (__bf16*)carve((size_t)MT1 * KC1 * 32 * 16 * 2); // 9.4 MB
    __bf16* xB   = (__bf16*)carve((size_t)NT2 * KC2 * 32 * 16 * 2); // 9.4 MB
    __bf16* wcb  = (__bf16*)carve((size_t)NT1 * KC1 * 32 * 16 * 2); // 3.0 MB
    float*  sbuf = (float*) carve((size_t)B_ * N1 * 4);             // 0.3 MB
    __bf16* vT   = (__bf16*)carve((size_t)N1 * B_ * 2);             // 0.2 MB
    float*  M2   = (float*) carve((size_t)N1 * K1 * 4);             // 5.9 MB
    float*  bij  = (float*) carve((size_t)I_ * O_ * 4);
    float*  cbuf = (float*) carve((size_t)I_ * O_ * 4);

    // one-time: zero logits, swizzle x into A- and B-tile layouts (bf16)
    zero_f32<<<(I_ * O_ + 255) / 256, 256, 0, stream>>>(bij, I_ * O_);
    convert_xA<<<(MT1 * KC1 * 32 + 255) / 256, 256, 0, stream>>>(x, xA);
    convert_xB<<<(NT2 * KC2 * 32 + 255) / 256, 256, 0, stream>>>(x, xB);

    for (int it = 0; it < 3; ++it) {
        softmax_c<<<1, 320, 0, stream>>>(bij, cbuf);
        build_wc<<<(NT1 * KC1 * 32 + 255) / 256, 256, 0, stream>>>(W, cbuf, wcb);
        zero_f32<<<(B_ * N1 + 255) / 256, 256, 0, stream>>>(sbuf, B_ * N1);
        gemm_s<<<dim3(MT1 / 8, NT1 / 2, KSPLIT), 256, 0, stream>>>(xA, wcb, sbuf);
        squash_k<<<(B_ * O_ + 255) / 256, 256, 0, stream>>>(sbuf, vT, out,
                                                            it == 2 ? 1 : 0);
        if (it < 2) {  // last iteration's b update never affects the output
            gemm_agree<<<dim3(NT2 / 2, NT1 / 2), 32, 0, stream>>>(vT, xB, M2);
            bupdate<<<(I_ * O_ + 255) / 256, 256, 0, stream>>>(W, M2, bij);
        }
    }
}
